// GAT_3315714752915
// MI455X (gfx1250) — matrix operand order repaired and checked by reading, not compiled, not run on hardware
//
#include <hip/hip_runtime.h>
#include <hip/hip_bf16.h>
#include <stdint.h>

typedef __bf16 bf16;
typedef __attribute__((ext_vector_type(16))) __bf16 v16bf;
typedef __attribute__((ext_vector_type(8)))  float  v8f;
typedef __attribute__((ext_vector_type(4)))  unsigned int v4u;

#define F_IN   128
#define H1     3
#define D1     192
#define D2     64
#define NCLS   10
#define NGRAPH 128

static inline int cdiv(long a, long b){ return (int)((a + b - 1) / b); }

// ---------------- generic helpers ----------------
__global__ void k_f32_to_bf16(const float* __restrict__ in, bf16* __restrict__ out, int n){
  int i = blockIdx.x * 256 + threadIdx.x;
  if (i < n) out[i] = (bf16)in[i];
}

__global__ void k_fill(float* __restrict__ p, float v, int n){
  int i = blockIdx.x * 256 + threadIdx.x;
  if (i < n) p[i] = v;
}

// ---------------- WMMA GEMM ----------------
// C[nrows,NCOLS](f32) = A[nrows,KDIM](bf16, row-major) * B[KDIM,NCOLS](bf16, row-major)
// 4 waves/block, 16 output rows per block, COLT 16x16 col-tiles per wave.
template<int KDIM, int NCOLS, int COLT>
__global__ __launch_bounds__(128) void k_gemm_wmma(const bf16* __restrict__ A,
                                                   const bf16* __restrict__ B,
                                                   float* __restrict__ C, int nrows){
  constexpr int LDK = KDIM + 8;                       // padded (bank-rotating) col stride
  __shared__ __align__(16) bf16 Blds[NCOLS * LDK];    // B transposed: [n][k]
  for (int i = threadIdx.x; i < KDIM * NCOLS; i += 128){
    int k = i / NCOLS, n = i - k * NCOLS;             // coalesced global read
    Blds[n * LDK + k] = B[i];
  }
  __syncthreads();

  const int lane = threadIdx.x & 31;
  const int wave = threadIdx.x >> 5;
  const int hi   = lane >> 4;                         // lane half
  const int lnm  = lane & 15;
  const int row0 = blockIdx.x * 16;
  int arow = row0 + lnm; if (arow >= nrows) arow = nrows - 1;

  v8f acc[COLT] = {};

  for (int k0 = 0; k0 < KDIM; k0 += 32){
    // A fragment, ISA 16-bit 16x32 layout:
    //   lanes 0-15 : M=lane,    K = {0..7} (v0-3) and {16..23} (v4-7)
    //   lanes 16-31: M=lane-16, K = {8..15}        and {24..31}
    union { v16bf v; unsigned int u[8]; } a;
    const unsigned int* ap = (const unsigned int*)(A + (size_t)arow * KDIM + k0);
    #pragma unroll
    for (int vi = 0; vi < 8; ++vi){
      int kg = vi >> 2, vv = vi & 3;
      a.u[vi] = ap[kg * 8 + hi * 4 + vv];             // dword = 2 contiguous K values
    }
    #pragma unroll
    for (int t = 0; t < COLT; ++t){
      int col0 = (wave * COLT + t) * 16;
      // B fragment: N = lane&15, K = k0 + 8*hi + {0..7} and k0 + 16 + 8*hi + {0..7}
      const v4u* bp = reinterpret_cast<const v4u*>(&Blds[(col0 + lnm) * LDK + k0 + hi * 8]);
      union { v16bf v; v4u q[2]; } b;
      b.q[0] = bp[0]; b.q[1] = bp[2];
      acc[t] = __builtin_amdgcn_wmma_f32_16x16x32_bf16(false, a.v, false, b.v,
                                                       (short)0, acc[t], false, false);
    }
  }
  #pragma unroll
  for (int t = 0; t < COLT; ++t){
    int col0 = (wave * COLT + t) * 16;
    #pragma unroll
    for (int r = 0; r < 8; ++r){                      // VGPR r -> M = r + 8*hi
      int row = row0 + r + 8 * hi;
      if (row < nrows) C[(size_t)row * NCOLS + col0 + lnm] = acc[t][r];
    }
  }
}

// ---------------- attention scores ----------------
template<int H, int C>
__global__ void k_att_scores(const float* __restrict__ xp,
                             const float* __restrict__ att_s,
                             const float* __restrict__ att_d,
                             float* __restrict__ as_, float* __restrict__ ad_, int N){
  int t = blockIdx.x * 256 + threadIdx.x;
  if (t >= N * H) return;
  int n = t / H, h = t - n * H;
  const float* row = xp + (size_t)n * (H * C) + h * C;
  const float* vs = att_s + h * C;
  const float* vd = att_d + h * C;
  float s = 0.f, d = 0.f;
  #pragma unroll 8
  for (int c = 0; c < C; ++c){ float x = row[c]; s += x * vs[c]; d += x * vd[c]; }
  as_[t] = s; ad_[t] = d;
}

// float atomic max via sign-split int trick (init to -3e38)
__device__ __forceinline__ void atomicMaxF(float* addr, float val){
  if (val >= 0.f) atomicMax((int*)addr, __float_as_int(val));
  else            atomicMin((unsigned int*)addr, __float_as_uint(val));
}

// ---------------- edge softmax passes ----------------
template<int H>
__global__ void k_edge_alpha_max(const int* __restrict__ src, const int* __restrict__ dst,
                                 int E, int Etot,
                                 const float* __restrict__ as_, const float* __restrict__ ad_,
                                 float* __restrict__ alpha, float* __restrict__ amax){
  int t = blockIdx.x * 256 + threadIdx.x;
  if (t >= Etot * H) return;
  int e = t / H, h = t - e * H;
  int s, d;
  if (e < E){ s = src[e]; d = dst[e]; } else { s = d = e - E; }   // self loops appended
  float a = as_[s * H + h] + ad_[d * H + h];
  a = (a > 0.f) ? a : 0.2f * a;                                   // leaky_relu 0.2
  alpha[t] = a;
  atomicMaxF(&amax[d * H + h], a);
}

template<int H>
__global__ void k_edge_exp_sum(const int* __restrict__ dst, int E, int Etot,
                               const float* __restrict__ amax, float* __restrict__ alpha,
                               float* __restrict__ denom){
  int t = blockIdx.x * 256 + threadIdx.x;
  if (t >= Etot * H) return;
  int e = t / H, h = t - e * H;
  int d = (e < E) ? dst[e] : (e - E);
  float ex = __expf(alpha[t] - amax[d * H + h]);
  alpha[t] = ex;                                                  // reuse buffer for exp
  atomicAdd(&denom[d * H + h], ex);
}

template<int H, int D>
__global__ void k_edge_aggregate(const int* __restrict__ src, const int* __restrict__ dst,
                                 int E, int Etot,
                                 const float* __restrict__ ex, const float* __restrict__ denom,
                                 const float* __restrict__ xp, float* __restrict__ agg){
  constexpr int CH4 = D / 4;
  long t = (long)blockIdx.x * 256 + threadIdx.x;
  if (t >= (long)Etot * CH4) return;
  int e = (int)(t / CH4), q = (int)(t - (long)e * CH4);
  int c = q * 4;
  int h = c / (D / H);
  int s, d;
  if (e < E){ s = src[e]; d = dst[e]; } else { s = d = e - E; }
  float coef = ex[e * H + h] / fmaxf(denom[d * H + h], 1e-16f);
  float4 v = *(const float4*)(xp + (size_t)s * D + c);
  float* o = agg + (size_t)d * D + c;
  atomicAdd(o + 0, v.x * coef);
  atomicAdd(o + 1, v.y * coef);
  atomicAdd(o + 2, v.z * coef);
  atomicAdd(o + 3, v.w * coef);
}

// ---------------- layer-1 epilogue: +bias, ELU, cast to bf16 ----------------
__global__ void k_bias_elu_bf16(const float* __restrict__ agg, const float* __restrict__ bias,
                                bf16* __restrict__ out, int n){
  int t = blockIdx.x * 256 + threadIdx.x;
  if (t >= n) return;
  float v = agg[t] + bias[t % D1];
  v = (v > 0.f) ? v : (__expf(v) - 1.f);
  out[t] = (bf16)v;
}

// ---------------- pooling + classifier ----------------
__global__ void k_count(const int* __restrict__ batch, float* __restrict__ cnt, int N){
  int i = blockIdx.x * 256 + threadIdx.x;
  if (i < N) atomicAdd(&cnt[batch[i]], 1.0f);
}

__global__ void k_pool_sum(const int* __restrict__ batch, const float* __restrict__ h,
                           float* __restrict__ pooled, int N){
  int t = blockIdx.x * 256 + threadIdx.x;
  if (t >= N * D2) return;
  int n = t / D2, c = t - n * D2;
  atomicAdd(&pooled[batch[n] * D2 + c], h[t]);
}

__global__ void k_final_linear(const float* __restrict__ pooled, const float* __restrict__ cnt,
                               const float* __restrict__ b2, const float* __restrict__ linW,
                               const float* __restrict__ linb, float* __restrict__ out){
  int t = blockIdx.x * 64 + threadIdx.x;
  if (t >= NGRAPH * NCLS) return;
  int g = t / NCLS, j = t - g * NCLS;
  float inv = 1.0f / fmaxf(cnt[g], 1.0f);
  float acc = linb[j];
  #pragma unroll 8
  for (int c = 0; c < D2; ++c){
    float m = pooled[g * D2 + c] * inv + b2[c];       // bias folded post-mean
    acc += m * linW[c * NCLS + j];
  }
  out[t] = acc;
}

// ---------------- driver ----------------
extern "C" void kernel_launch(void* const* d_in, const int* in_sizes, int n_in,
                              void* d_out, int out_size, void* d_ws, size_t ws_size,
                              hipStream_t stream) {
  const float* x      = (const float*)d_in[0];
  const int*   ei     = (const int*)  d_in[1];
  const int*   batch  = (const int*)  d_in[2];
  const float* W1     = (const float*)d_in[3];
  const float* atts1  = (const float*)d_in[4];
  const float* attd1  = (const float*)d_in[5];
  const float* b1     = (const float*)d_in[6];
  const float* W2     = (const float*)d_in[7];
  const float* atts2  = (const float*)d_in[8];
  const float* attd2  = (const float*)d_in[9];
  const float* b2     = (const float*)d_in[10];
  const float* linW   = (const float*)d_in[11];
  const float* linb   = (const float*)d_in[12];

  const int N    = in_sizes[0] / F_IN;
  const int E    = in_sizes[1] / 2;
  const int Etot = E + N;                 // edges + self-loops
  const int* srcI = ei;                   // edge_index[0]
  const int* dstI = ei + E;               // edge_index[1]

  // ---- workspace carve (256B aligned) ----
  char* w = (char*)d_ws;
  auto carve = [&](size_t bytes)->void*{ void* p = w; w += (bytes + 255) & ~(size_t)255; return p; };
  bf16*  xbf    = (bf16*) carve((size_t)N * F_IN * 2);
  bf16*  w1bf   = (bf16*) carve((size_t)F_IN * D1 * 2);
  bf16*  w2bf   = (bf16*) carve((size_t)D1 * D2 * 2);
  float* xp1    = (float*)carve((size_t)N * D1 * 4);
  float* as1    = (float*)carve((size_t)N * H1 * 4);
  float* ad1    = (float*)carve((size_t)N * H1 * 4);
  float* amax1  = (float*)carve((size_t)N * H1 * 4);
  float* den1   = (float*)carve((size_t)N * H1 * 4);
  float* alpha1 = (float*)carve((size_t)Etot * H1 * 4);
  float* agg1   = (float*)carve((size_t)N * D1 * 4);
  bf16*  h1bf   = (bf16*) carve((size_t)N * D1 * 2);
  float* xp2    = (float*)carve((size_t)N * D2 * 4);
  float* as2    = (float*)carve((size_t)N * 4);
  float* ad2    = (float*)carve((size_t)N * 4);
  float* amax2  = (float*)carve((size_t)N * 4);
  float* den2   = (float*)carve((size_t)N * 4);
  float* alpha2 = (float*)carve((size_t)Etot * 4);
  float* agg2   = (float*)carve((size_t)N * D2 * 4);
  float* pooled = (float*)carve((size_t)NGRAPH * D2 * 4);
  float* cnt    = (float*)carve((size_t)NGRAPH * 4);

  // ---- precision conversion for WMMA ----
  k_f32_to_bf16<<<cdiv((long)N * F_IN, 256), 256, 0, stream>>>(x,  xbf,  N * F_IN);
  k_f32_to_bf16<<<cdiv(F_IN * D1, 256),     256, 0, stream>>>(W1, w1bf, F_IN * D1);
  k_f32_to_bf16<<<cdiv(D1 * D2, 256),       256, 0, stream>>>(W2, w2bf, D1 * D2);

  // ================= layer 1 (H=3, C=64, concat) =================
  k_gemm_wmma<F_IN, D1, 3><<<cdiv(N, 16), 128, 0, stream>>>(xbf, w1bf, xp1, N);
  k_att_scores<H1, 64><<<cdiv((long)N * H1, 256), 256, 0, stream>>>(xp1, atts1, attd1, as1, ad1, N);

  k_fill<<<cdiv((long)N * H1, 256), 256, 0, stream>>>(amax1, -3.0e38f, N * H1);
  k_fill<<<cdiv((long)N * H1, 256), 256, 0, stream>>>(den1, 0.f, N * H1);
  k_edge_alpha_max<H1><<<cdiv((long)Etot * H1, 256), 256, 0, stream>>>(srcI, dstI, E, Etot, as1, ad1, alpha1, amax1);
  k_edge_exp_sum<H1><<<cdiv((long)Etot * H1, 256), 256, 0, stream>>>(dstI, E, Etot, amax1, alpha1, den1);

  k_fill<<<cdiv((long)N * D1, 256), 256, 0, stream>>>(agg1, 0.f, N * D1);
  k_edge_aggregate<H1, D1><<<cdiv((long)Etot * (D1 / 4), 256), 256, 0, stream>>>(srcI, dstI, E, Etot, alpha1, den1, xp1, agg1);
  k_bias_elu_bf16<<<cdiv((long)N * D1, 256), 256, 0, stream>>>(agg1, b1, h1bf, N * D1);

  // ================= layer 2 (H=1, C=64, mean==identity) =================
  k_gemm_wmma<D1, D2, 1><<<cdiv(N, 16), 128, 0, stream>>>(h1bf, w2bf, xp2, N);
  k_att_scores<1, D2><<<cdiv(N, 256), 256, 0, stream>>>(xp2, atts2, attd2, as2, ad2, N);

  k_fill<<<cdiv(N, 256), 256, 0, stream>>>(amax2, -3.0e38f, N);
  k_fill<<<cdiv(N, 256), 256, 0, stream>>>(den2, 0.f, N);
  k_edge_alpha_max<1><<<cdiv(Etot, 256), 256, 0, stream>>>(srcI, dstI, E, Etot, as2, ad2, alpha2, amax2);
  k_edge_exp_sum<1><<<cdiv(Etot, 256), 256, 0, stream>>>(dstI, E, Etot, amax2, alpha2, den2);

  k_fill<<<cdiv((long)N * D2, 256), 256, 0, stream>>>(agg2, 0.f, N * D2);
  k_edge_aggregate<1, D2><<<cdiv((long)Etot * (D2 / 4), 256), 256, 0, stream>>>(srcI, dstI, E, Etot, alpha2, den2, xp2, agg2);

  // ================= mean pool + classifier (b2 folded in) =================
  k_fill<<<cdiv(NGRAPH * D2, 256), 256, 0, stream>>>(pooled, 0.f, NGRAPH * D2);
  k_fill<<<1, 256, 0, stream>>>(cnt, 0.f, NGRAPH);
  k_count<<<cdiv(N, 256), 256, 0, stream>>>(batch, cnt, N);
  k_pool_sum<<<cdiv((long)N * D2, 256), 256, 0, stream>>>(batch, agg2, pooled, N);
  k_final_linear<<<cdiv(NGRAPH * NCLS, 64), 64, 0, stream>>>(pooled, cnt, b2, linW, linb, (float*)d_out);
}
